// ResRnn_35905926594981
// MI455X (gfx1250) — matrix operand order repaired and checked by reading, not compiled, not run on hardware
//
#include <hip/hip_runtime.h>

#define SEQ      512
#define BATCH    256
#define IW       64
#define STREAM   1024
#define OUTW     128
#define NWG      64
#define NTHR     256
#define CPW      16      // columns per workgroup
#define WAVES    8

typedef __attribute__((ext_vector_type(16))) __bf16   v16bf;
typedef __attribute__((ext_vector_type(8)))  float    v8f;
typedef __attribute__((ext_vector_type(4)))  unsigned v4u;

union Frag { v4u u[2]; v16bf v; };

__device__ __forceinline__ float huber(float x) {
    float d = fabsf(x);
    return d < 1.0f ? 0.5f * d * d : d - 0.5f;
}

__device__ __forceinline__ void grid_barrier(unsigned* cnt, unsigned* gen) {
    __threadfence();                 // release: flush this thread's stores to device scope
    __syncthreads();
    if (threadIdx.x == 0) {
        unsigned g = __hip_atomic_load(gen, __ATOMIC_ACQUIRE, __HIP_MEMORY_SCOPE_AGENT);
        unsigned a = __hip_atomic_fetch_add(cnt, 1u, __ATOMIC_ACQ_REL, __HIP_MEMORY_SCOPE_AGENT);
        if (a == NWG - 1u) {
            __hip_atomic_store(cnt, 0u, __ATOMIC_RELAXED, __HIP_MEMORY_SCOPE_AGENT);
            __hip_atomic_fetch_add(gen, 1u, __ATOMIC_RELEASE, __HIP_MEMORY_SCOPE_AGENT);
        } else {
            while (__hip_atomic_load(gen, __ATOMIC_ACQUIRE, __HIP_MEMORY_SCOPE_AGENT) == g)
                __builtin_amdgcn_s_sleep(8);
        }
    }
    __syncthreads();
    __threadfence();                 // acquire: invalidate per-WGP caches before re-reading s/h
}

// LDS B-fragment: weights stored in fragment order: [(k>>4)][col_local][k&15]
__device__ __forceinline__ v16bf load_bfrag_lds(const __bf16* Wl, int kk, int half, int l15) {
    Frag f;
    const v4u* p = (const v4u*)(Wl + (((kk * 2 + half) * 16 + l15) << 4));
    f.u[0] = p[0];
    f.u[1] = p[1];
    return f.v;
}

// Global A-fragment (row-major bf16): lane holds K = k0+half*16+{0..15}
__device__ __forceinline__ v16bf load_afrag(const __bf16* A, int row, int k0, int half) {
    Frag f;
    const __bf16* base = A + row * STREAM + k0 + half * 16;
    f.u[0] = *(const v4u*)(base);
    f.u[1] = *(const v4u*)(base + 8);
    return f.v;
}

extern "C" __global__ __launch_bounds__(NTHR)
void resrnn_persistent(const float* __restrict__ in,  const float* __restrict__ ist,
                       const float* __restrict__ affw, const float* __restrict__ affb,
                       const float* __restrict__ w1,  const float* __restrict__ b1,
                       const float* __restrict__ w2,  const float* __restrict__ b2,
                       float* __restrict__ out,
                       float* s32, __bf16* sbf, __bf16* hbf,
                       float* partials, unsigned* barCnt, unsigned* barGen) {
    extern __shared__ __bf16 smem[];            // 2 * 16*1024 bf16 = 64 KiB
    __bf16* W1s = smem;
    __bf16* W2s = smem + CPW * STREAM;

    const int tid  = threadIdx.x;
    const int wg   = blockIdx.x;
    const int colBase = wg * CPW;
    const int w    = tid >> 5;
    const int lane = tid & 31;
    const int half = lane >> 4;
    const int l15  = lane & 15;

    // ---- convert weight slices to bf16, fragment order, into LDS (once) ----
    for (int idx = tid; idx < CPW * STREAM; idx += NTHR) {
        int k  = idx & (STREAM - 1);
        int cl = idx >> 10;
        int li = ((k >> 4) << 8) + (cl << 4) + (k & 15);
        W1s[li] = (__bf16)w1[(colBase + cl) * STREAM + k];
        W2s[li] = (__bf16)w2[(colBase + cl) * STREAM + k];
    }

    // ---- init s0 = aff( concat(input[0], init_state[:,64:]) ) for own columns ----
    for (int idx = tid; idx < BATCH * CPW; idx += NTHR) {
        int cl = idx & 15, r = idx >> 4;
        int c = colBase + cl;
        float v = (c < IW) ? in[r * IW + c] : ist[r * STREAM + c];
        v = affw[c] * v + affb[c];
        s32[r * STREAM + c] = v;
        sbf[r * STREAM + c] = (__bf16)v;
    }

    const float b1v = b1[colBase + l15];
    const float b2v = b2[colBase + l15];
    const float Pc = 0.97f, Qc = 1.0f - 0.97f;

    grid_barrier(barCnt, barGen);

    const int rowA0 = 32 * w + l15;
    const int rowA1 = rowA0 + 16;
    const int statCol = colBase + 2 * w + half;   // each wave-half owns one column

    for (int t = 0; t < SEQ; ++t) {
        // ================= phase 1: reg stats + h = |s @ W1^T + b1| =================
        {
            float sum = 0.f, sq = 0.f;
            const float* sc = s32 + statCol;
            #pragma unroll 4
            for (int i = 0; i < 16; ++i) {
                float v = sc[(l15 * 16 + i) * STREAM];
                sum += v; sq += v * v;
            }
            #pragma unroll
            for (int m = 8; m >= 1; m >>= 1) {
                sum += __shfl_xor(sum, m, 32);
                sq  += __shfl_xor(sq,  m, 32);
            }
            float mean = sum * (1.0f / BATCH);
            float var  = (sq - sum * mean) * (1.0f / (BATCH - 1));
            float contrib = huber(mean) + huber(var - 1.0f);
            float other = __shfl(contrib, 16, 32);     // lane16's column
            if (lane == 0)
                partials[(t * NWG + wg) * WAVES + w] = contrib + other;
        }
        {
            v8f acc0 = {}; v8f acc1 = {};
            #pragma unroll 4
            for (int kk = 0; kk < 32; ++kk) {
                v16bf bf = load_bfrag_lds(W1s, kk, half, l15);
                v16bf a0 = load_afrag(sbf, rowA0, kk * 32, half);
                v16bf a1 = load_afrag(sbf, rowA1, kk * 32, half);
                acc0 = __builtin_amdgcn_wmma_f32_16x16x32_bf16(false, a0, false, bf, (short)0, acc0, false, false);
                acc1 = __builtin_amdgcn_wmma_f32_16x16x32_bf16(false, a1, false, bf, (short)0, acc1, false, false);
            }
            int col = colBase + l15;
            int r0  = 32 * w + half * 8;
            #pragma unroll
            for (int r = 0; r < 8; ++r) {
                hbf[(r0 + r)      * STREAM + col] = (__bf16)fabsf(acc0[r] + b1v);
                hbf[(r0 + 16 + r) * STREAM + col] = (__bf16)fabsf(acc1[r] + b1v);
            }
        }
        grid_barrier(barCnt, barGen);

        // ======== phase 2: s' = P*s + (1-P)*(h @ W2^T + b2); inject input[t+1] ========
        if (colBase >= IW) {
            v8f acc0 = {}; v8f acc1 = {};
            #pragma unroll 4
            for (int kk = 0; kk < 32; ++kk) {
                v16bf bf = load_bfrag_lds(W2s, kk, half, l15);
                v16bf a0 = load_afrag(hbf, rowA0, kk * 32, half);
                v16bf a1 = load_afrag(hbf, rowA1, kk * 32, half);
                acc0 = __builtin_amdgcn_wmma_f32_16x16x32_bf16(false, a0, false, bf, (short)0, acc0, false, false);
                acc1 = __builtin_amdgcn_wmma_f32_16x16x32_bf16(false, a1, false, bf, (short)0, acc1, false, false);
            }
            int col = colBase + l15;
            int r0  = 32 * w + half * 8;
            #pragma unroll
            for (int r = 0; r < 8; ++r) {
                int row0 = r0 + r;
                float nv0 = Pc * s32[row0 * STREAM + col] + Qc * (acc0[r] + b2v);
                s32[row0 * STREAM + col] = nv0;
                sbf[row0 * STREAM + col] = (__bf16)nv0;
                int row1 = r0 + 16 + r;
                float nv1 = Pc * s32[row1 * STREAM + col] + Qc * (acc1[r] + b2v);
                s32[row1 * STREAM + col] = nv1;
                sbf[row1 * STREAM + col] = (__bf16)nv1;
            }
        } else if (t + 1 < SEQ) {
            // columns 0..63 are always replaced by the next input; skip the GEMM
            const float* nxt = in + (size_t)(t + 1) * BATCH * IW;
            for (int idx = tid; idx < BATCH * CPW; idx += NTHR) {
                int cl = idx & 15, r = idx >> 4;
                int c = colBase + cl;               // < 64
                float v = nxt[r * IW + c];
                s32[r * STREAM + c] = v;
                sbf[r * STREAM + c] = (__bf16)v;
            }
        }
        grid_barrier(barCnt, barGen);
    }

    // ================= epilogue =================
    if (wg == 0) {
        float sum = 0.f;
        for (int i = tid; i < SEQ * NWG * WAVES; i += NTHR) sum += partials[i];
        float* red = (float*)smem;                 // reuse LDS
        red[tid] = sum;
        __syncthreads();
        for (int s = NTHR / 2; s > 0; s >>= 1) {
            if (tid < s) red[tid] += red[tid + s];
            __syncthreads();
        }
        if (tid == 0) out[BATCH * OUTW] = red[0] * (1.0f / (1024.0f * 512.0f));
    }
    if (colBase >= STREAM - OUTW) {
        for (int idx = tid; idx < BATCH * CPW; idx += NTHR) {
            int cl = idx & 15, r = idx >> 4;
            int c = colBase + cl;
            out[r * OUTW + (c - (STREAM - OUTW))] = s32[r * STREAM + c];
        }
    }
}

extern "C" void kernel_launch(void* const* d_in, const int* in_sizes, int n_in,
                              void* d_out, int out_size, void* d_ws, size_t ws_size,
                              hipStream_t stream) {
    const float* in   = (const float*)d_in[0];
    const float* ist  = (const float*)d_in[1];
    const float* affw = (const float*)d_in[2];
    const float* affb = (const float*)d_in[3];
    const float* w1   = (const float*)d_in[4];
    const float* b1   = (const float*)d_in[5];
    const float* w2   = (const float*)d_in[6];
    const float* b2   = (const float*)d_in[7];
    float* out = (float*)d_out;

    char* ws = (char*)d_ws;
    float*    s32      = (float*)(ws);                               // 1 MiB
    __bf16*   sbf      = (__bf16*)(ws + (1u << 20));                 // 512 KiB
    __bf16*   hbf      = (__bf16*)(ws + (1u << 20) + (512u << 10));  // 512 KiB
    float*    partials = (float*)(ws + (2u << 20));                  // 512*64*8*4 = 1 MiB
    unsigned* bar      = (unsigned*)(ws + (3u << 20));               // cnt, gen

    hipMemsetAsync(bar, 0, 2 * sizeof(unsigned), stream);

    size_t shmem = (size_t)2 * CPW * STREAM * 2;                     // 65536 B
    hipLaunchKernelGGL(resrnn_persistent, dim3(NWG), dim3(NTHR), shmem, stream,
                       in, ist, affw, affb, w1, b1, w2, b2, out,
                       s32, sbf, hbf, partials, bar, bar + 1);
}
